// TransformerBlock_13048110645954
// MI455X (gfx1250) — compile-verified, matrix operand order checked by reading, not run on hardware
//
#include <hip/hip_runtime.h>
#include <hip/hip_bf16.h>
#include <math.h>

// ---------------- types / helpers ----------------
typedef __attribute__((ext_vector_type(16))) __bf16 v16bf;
typedef __attribute__((ext_vector_type(8)))  float  v8f;
typedef __attribute__((ext_vector_type(4)))  unsigned int u32x4;

union FragBF { u32x4 q[2]; v16bf v; };

__device__ inline unsigned int f32_bf16(float f) {
  unsigned int u = __float_as_uint(f);
  unsigned int r = 0x7FFFu + ((u >> 16) & 1u);
  return (u + r) >> 16;
}
// pack 8 consecutive values (float->bf16 or passthrough bf16 bits) into u32x4
__device__ inline u32x4 load8_bf(const float* p) {
  const float4* q = (const float4*)p;
  float4 a = q[0], b = q[1];
  u32x4 r;
  r.x = f32_bf16(a.x) | (f32_bf16(a.y) << 16);
  r.y = f32_bf16(a.z) | (f32_bf16(a.w) << 16);
  r.z = f32_bf16(b.x) | (f32_bf16(b.y) << 16);
  r.w = f32_bf16(b.z) | (f32_bf16(b.w) << 16);
  return r;
}
__device__ inline u32x4 load8_bf(const unsigned short* p) {
  return *(const u32x4*)p;
}

// DPP16 butterfly reductions over a 16-lane row (VALU only, no LDS)
// ctrl must be a compile-time constant -> template parameter
template <int CTRL>
__device__ inline float dpp_f(float x) {
  return __int_as_float(
      __builtin_amdgcn_update_dpp(0, __float_as_int(x), CTRL, 0xF, 0xF, true));
}
#define DPP_QPERM_XOR1 0xB1   // quad_perm(1,0,3,2)
#define DPP_QPERM_XOR2 0x4E   // quad_perm(2,3,0,1)
#define DPP_ROW_HMIRR  0x141  // row_half_mirror (lane^7 within 8)
#define DPP_ROW_MIRR   0x140  // row_mirror (lane^15 within 16)
__device__ inline float red16_max(float x) {
  x = fmaxf(x, dpp_f<DPP_QPERM_XOR1>(x));
  x = fmaxf(x, dpp_f<DPP_QPERM_XOR2>(x));
  x = fmaxf(x, dpp_f<DPP_ROW_HMIRR>(x));
  x = fmaxf(x, dpp_f<DPP_ROW_MIRR>(x));
  return x;
}
__device__ inline float red16_sum(float x) {
  x += dpp_f<DPP_QPERM_XOR1>(x);
  x += dpp_f<DPP_QPERM_XOR2>(x);
  x += dpp_f<DPP_ROW_HMIRR>(x);
  x += dpp_f<DPP_ROW_MIRR>(x);
  return x;
}

#define NTOK 4096   // B*T
#define CDIM 1024
#define IDIM 4096
#define NEXP 8

// ---------------- RMSNorm ----------------
__global__ void rmsnorm_kernel(const float* __restrict__ x,
                               const float* __restrict__ scale,
                               float* __restrict__ out) {
  __shared__ float red[9];
  int row = blockIdx.x;
  int tid = threadIdx.x;
  const float* xr = x + (size_t)row * CDIM;
  float v[4]; float ss = 0.f;
#pragma unroll
  for (int i = 0; i < 4; i++) { v[i] = xr[tid + 256 * i]; ss += v[i] * v[i]; }
#pragma unroll
  for (int off = 16; off >= 1; off >>= 1) ss += __shfl_xor(ss, off);
  if ((tid & 31) == 0) red[tid >> 5] = ss;
  __syncthreads();
  if (tid == 0) {
    float t = 0.f;
#pragma unroll
    for (int w = 0; w < 8; w++) t += red[w];
    red[8] = t;
  }
  __syncthreads();
  float inv = rsqrtf(red[8] / (float)CDIM + 1e-5f);
#pragma unroll
  for (int i = 0; i < 4; i++) {
    int c = tid + 256 * i;
    out[(size_t)row * CDIM + c] = v[i] * inv * scale[c];
  }
}

// ---------------- generic WMMA GEMM, 128x128 tile, double-buffered LDS ----------------
// C(M,N) = A(M,K) * B(K,N); A: float or bf16-bits; B: float (bf16-converted on stage)
// 8 waves; wave (wr,wc) owns rows wr*32..+32, cols wc*64..+64 => 2x4 wmma accums.
// EPI: 0 = store f32; 1 = gelu->bf16 (outh); 2 = +aux residual; 3 = outf += aux[r*8+e]*acc
template <int EPI, typename AT>
__global__ void gemm_wmma(const AT* __restrict__ A, const float* __restrict__ Bm,
                          float* __restrict__ outf, unsigned short* __restrict__ outh,
                          const float* __restrict__ aux,
                          int M, int N, int Kd, int lda, int ldb, int ldc, int expert) {
  __shared__ __attribute__((aligned(16))) unsigned short aS[2][128][40];
  __shared__ __attribute__((aligned(16))) unsigned short bS[2][128][40];
  int tid = threadIdx.x;
  int lane = tid & 31;
  int wave = tid >> 5;
  int wr = wave & 3, wc = wave >> 2;
  int m0 = blockIdx.y * 128, n0 = blockIdx.x * 128;
  int hi = lane >> 4;
  int kb = hi * 8;
  int ln = lane & 15;
  int ar0 = wr * 32 + ln;
  // staging index precompute (2 chunks of 8 elems per thread for A and B)
  int rA[2], cgA[2], kB[2], nbB[2];
#pragma unroll
  for (int g = 0; g < 2; g++) {
    int lin = tid + 256 * g;
    rA[g] = lin >> 2; cgA[g] = (lin & 3) * 8;
    kB[g] = lin >> 4; nbB[g] = (lin & 15) * 8;
  }
  v8f acc[2][4] = {};
  u32x4 ra[2], rb[2];
  int nt = Kd >> 5;
  // prologue: load tile 0 into regs, store to buffer 0
#pragma unroll
  for (int g = 0; g < 2; g++) {
    ra[g] = load8_bf(&A[(size_t)(m0 + rA[g]) * lda + cgA[g]]);
    const float4* qp = (const float4*)&Bm[(size_t)kB[g] * ldb + n0 + nbB[g]];
    float4 fa = qp[0], fb = qp[1];
    rb[g].x = f32_bf16(fa.x) | (f32_bf16(fa.y) << 16);
    rb[g].y = f32_bf16(fa.z) | (f32_bf16(fa.w) << 16);
    rb[g].z = f32_bf16(fb.x) | (f32_bf16(fb.y) << 16);
    rb[g].w = f32_bf16(fb.z) | (f32_bf16(fb.w) << 16);
  }
#pragma unroll
  for (int g = 0; g < 2; g++) {
    *(u32x4*)&aS[0][rA[g]][cgA[g]] = ra[g];
#pragma unroll
    for (int j = 0; j < 8; j++) {
      unsigned w = ((const unsigned*)&rb[g])[j >> 1];
      bS[0][nbB[g] + j][kB[g]] = (unsigned short)((j & 1) ? (w >> 16) : (w & 0xFFFFu));
    }
  }
  __syncthreads();
  for (int kt = 0; kt < nt; kt++) {
    int buf = kt & 1;
    int k1 = (kt + 1) << 5;
    if (kt + 1 < nt) {
      // issue next tile's global loads (packed in regs)
#pragma unroll
      for (int g = 0; g < 2; g++) {
        ra[g] = load8_bf(&A[(size_t)(m0 + rA[g]) * lda + k1 + cgA[g]]);
        const float4* qp = (const float4*)&Bm[(size_t)(k1 + kB[g]) * ldb + n0 + nbB[g]];
        float4 fa = qp[0], fb = qp[1];
        rb[g].x = f32_bf16(fa.x) | (f32_bf16(fa.y) << 16);
        rb[g].y = f32_bf16(fa.z) | (f32_bf16(fa.w) << 16);
        rb[g].z = f32_bf16(fb.x) | (f32_bf16(fb.y) << 16);
        rb[g].w = f32_bf16(fb.z) | (f32_bf16(fb.w) << 16);
      }
    }
    if (kt + 2 < nt) {   // prefetch two tiles ahead into GL2
      int k2 = (kt + 2) << 5;
      __builtin_prefetch(&A[(size_t)(m0 + rA[0]) * lda + k2 + cgA[0]], 0, 1);
      __builtin_prefetch(&Bm[(size_t)(k2 + kB[0]) * ldb + n0 + nbB[0]], 0, 1);
    }
    // compute current tile
    FragBF a0, a1;
    a0.q[0] = *(const u32x4*)&aS[buf][ar0][kb];
    a0.q[1] = *(const u32x4*)&aS[buf][ar0][16 + kb];
    a1.q[0] = *(const u32x4*)&aS[buf][ar0 + 16][kb];
    a1.q[1] = *(const u32x4*)&aS[buf][ar0 + 16][16 + kb];
#pragma unroll
    for (int bj = 0; bj < 4; bj++) {
      FragBF bf;
      int bc = wc * 64 + bj * 16 + ln;
      bf.q[0] = *(const u32x4*)&bS[buf][bc][kb];
      bf.q[1] = *(const u32x4*)&bS[buf][bc][16 + kb];
      acc[0][bj] = __builtin_amdgcn_wmma_f32_16x16x32_bf16(false, a0.v, false, bf.v, (short)0, acc[0][bj], false, false);
      acc[1][bj] = __builtin_amdgcn_wmma_f32_16x16x32_bf16(false, a1.v, false, bf.v, (short)0, acc[1][bj], false, false);
    }
    if (kt + 1 < nt) {
      int nb = buf ^ 1;
#pragma unroll
      for (int g = 0; g < 2; g++) {
        *(u32x4*)&aS[nb][rA[g]][cgA[g]] = ra[g];
#pragma unroll
        for (int j = 0; j < 8; j++) {
          unsigned w = ((const unsigned*)&rb[g])[j >> 1];
          bS[nb][nbB[g] + j][kB[g]] = (unsigned short)((j & 1) ? (w >> 16) : (w & 0xFFFFu));
        }
      }
    }
    __syncthreads();
  }
#pragma unroll
  for (int ai = 0; ai < 2; ai++) {
#pragma unroll
    for (int bj = 0; bj < 4; bj++) {
#pragma unroll
      for (int i = 0; i < 8; i++) {
        int r = m0 + wr * 32 + ai * 16 + i + hi * 8;
        int c = n0 + wc * 64 + bj * 16 + ln;
        size_t idx = (size_t)r * ldc + c;
        float v = acc[ai][bj][i];
        if (EPI == 0) {
          outf[idx] = v;
        } else if (EPI == 1) {
          float g = 0.5f * v * (1.0f + erff(v * 0.70710678118f));
          outh[idx] = (unsigned short)f32_bf16(g);
        } else if (EPI == 2) {
          outf[idx] = v + aux[idx];
        } else {
          outf[idx] += aux[(size_t)r * NEXP + expert] * v;
        }
      }
    }
  }
}

// ---------------- RoPE (+ bf16 convert) ----------------
__global__ void rope_kernel(const float* __restrict__ in, unsigned short* __restrict__ outb,
                            int nheads, int ld) {
  int idx = blockIdx.x * 256 + threadIdx.x;       // pair index over NTOK*nheads*32
  int d = idx & 31;
  int th = idx >> 5;
  int head = th % nheads;
  int tok = th / nheads;
  int t = tok & 1023;                             // tok = b*T + t
  float inv_freq = __expf(-0.2878231366f * (float)d);  // 10000^(-d/32)
  float ang = (float)t * inv_freq;
  float cs = __cosf(ang), sn = __sinf(ang);
  size_t base = (size_t)tok * ld + head * 64;
  float x1 = in[base + d], x2 = in[base + 32 + d];
  outb[base + d]      = (unsigned short)f32_bf16(x1 * cs - x2 * sn);
  outb[base + 32 + d] = (unsigned short)f32_bf16(x2 * cs + x1 * sn);
}

__global__ void cvt_bf16_kernel(const float* __restrict__ in, unsigned short* __restrict__ outb) {
  size_t i = (size_t)blockIdx.x * 256 + threadIdx.x;
  outb[i] = (unsigned short)f32_bf16(in[i]);
}

// ---------------- flash attention (WMMA, causal, GQA) ----------------
__global__ void attn_kernel(const unsigned short* __restrict__ qb,
                            const unsigned short* __restrict__ kbuf,
                            const unsigned short* __restrict__ vbuf,
                            float* __restrict__ y) {
  __shared__ __attribute__((aligned(16))) unsigned short Kls[32][72];
  __shared__ __attribute__((aligned(16))) unsigned short Vls[64][40];
  __shared__ __attribute__((aligned(16))) unsigned short Pls[8][16][40];
  int tid = threadIdx.x, lane = tid & 31, wave = tid >> 5;
  int bh = blockIdx.x;
  int b = bh >> 4, h = bh & 15;
  int kvh = h >> 2;
  int qt = blockIdx.y;
  int qbase = qt * 128 + wave * 16;
  int hi = lane >> 4, kbo = hi * 8, ln = lane & 15;
  int qrow = qbase + ln;
  const unsigned short* qp = qb + (size_t)(b * 1024 + qrow) * CDIM + h * 64;
  FragBF qa0, qa1;
  qa0.q[0] = *(const u32x4*)(qp + kbo);
  qa0.q[1] = *(const u32x4*)(qp + 16 + kbo);
  qa1.q[0] = *(const u32x4*)(qp + 32 + kbo);
  qa1.q[1] = *(const u32x4*)(qp + 48 + kbo);
  float m[8], l[8];
  v8f o[4] = {};
#pragma unroll
  for (int i = 0; i < 8; i++) { m[i] = -3.0e38f; l[i] = 0.f; }
  int ntiles = (qt + 1) * 4;
  for (int kt = 0; kt < ntiles; kt++) {
    int key0 = kt * 32;
#pragma unroll
    for (int i = 0; i < 4; i++) {                 // stage K rows (32 x 64), uint copies
      int lin = tid + 256 * i;
      int kk = lin >> 5, du = lin & 31;
      const unsigned int* src =
          (const unsigned int*)(kbuf + (size_t)(b * 1024 + key0 + kk) * 256 + kvh * 64);
      ((unsigned int*)&Kls[kk][0])[du] = src[du];
    }
#pragma unroll
    for (int i = 0; i < 8; i++) {                 // stage V transposed: Vls[d][kk]
      int lin = tid + 256 * i;
      int kk = lin >> 6, d = lin & 63;
      Vls[d][kk] = vbuf[(size_t)(b * 1024 + key0 + kk) * 256 + kvh * 64 + d];
    }
    __syncthreads();
    v8f s[2] = {};
#pragma unroll
    for (int sub = 0; sub < 2; sub++) {           // scores: Q(16x64) * K^T(64x32)
      int krow = sub * 16 + ln;
      FragBF bK;
      bK.q[0] = *(const u32x4*)&Kls[krow][kbo];
      bK.q[1] = *(const u32x4*)&Kls[krow][16 + kbo];
      s[sub] = __builtin_amdgcn_wmma_f32_16x16x32_bf16(false, qa0.v, false, bK.v, (short)0, s[sub], false, false);
      bK.q[0] = *(const u32x4*)&Kls[krow][32 + kbo];
      bK.q[1] = *(const u32x4*)&Kls[krow][48 + kbo];
      s[sub] = __builtin_amdgcn_wmma_f32_16x16x32_bf16(false, qa1.v, false, bK.v, (short)0, s[sub], false, false);
    }
    int keyc0 = key0 + ln, keyc1 = key0 + 16 + ln;
#pragma unroll
    for (int i = 0; i < 8; i++) {                 // online softmax (DPP reductions)
      int row = qbase + i + hi * 8;
      float s0 = s[0][i] * 0.125f; if (keyc0 > row) s0 = -3.0e38f;
      float s1 = s[1][i] * 0.125f; if (keyc1 > row) s1 = -3.0e38f;
      float t = red16_max(fmaxf(s0, s1));
      float mn = fmaxf(m[i], t);
      float fac = __expf(m[i] - mn);
      float p0 = __expf(s0 - mn), p1 = __expf(s1 - mn);
      float ps = red16_sum(p0 + p1);
      l[i] = l[i] * fac + ps;
      m[i] = mn;
#pragma unroll
      for (int sub = 0; sub < 4; sub++) o[sub][i] *= fac;
      Pls[wave][i + hi * 8][ln]      = (unsigned short)f32_bf16(p0);
      Pls[wave][i + hi * 8][16 + ln] = (unsigned short)f32_bf16(p1);
    }
    __syncthreads();
    FragBF pa;                                    // P(16x32) * V(32x64)
    pa.q[0] = *(const u32x4*)&Pls[wave][ln][kbo];
    pa.q[1] = *(const u32x4*)&Pls[wave][ln][16 + kbo];
#pragma unroll
    for (int sub = 0; sub < 4; sub++) {
      FragBF bV;
      int col = sub * 16 + ln;
      bV.q[0] = *(const u32x4*)&Vls[col][kbo];
      bV.q[1] = *(const u32x4*)&Vls[col][16 + kbo];
      o[sub] = __builtin_amdgcn_wmma_f32_16x16x32_bf16(false, pa.v, false, bV.v, (short)0, o[sub], false, false);
    }
    __syncthreads();
  }
#pragma unroll
  for (int sub = 0; sub < 4; sub++)
#pragma unroll
    for (int i = 0; i < 8; i++) {
      int row = qbase + i + hi * 8;
      int col = h * 64 + sub * 16 + ln;
      y[(size_t)(b * 1024 + row) * CDIM + col] = o[sub][i] / l[i];
    }
}

// ---------------- MoE gate: softmax, top-2, combine weights, aux stats ----------------
__global__ void gate_kernel(const float* __restrict__ xn, const float* __restrict__ Wg,
                            float* __restrict__ cw, float* __restrict__ accum) {
  int tid = threadIdx.x, lane = tid & 31, wave = tid >> 5;
  int row = blockIdx.x * 8 + wave;
  float part[NEXP];
#pragma unroll
  for (int e = 0; e < NEXP; e++) part[e] = 0.f;
  const float* xr = xn + (size_t)row * CDIM;
  for (int c = lane; c < CDIM; c += 32) {
    float xv = xr[c];
#pragma unroll
    for (int e = 0; e < NEXP; e++) part[e] += xv * Wg[c * NEXP + e];
  }
#pragma unroll
  for (int e = 0; e < NEXP; e++)
#pragma unroll
    for (int off = 1; off < 32; off <<= 1) part[e] += __shfl_xor(part[e], off);
  if (lane == 0) {
    float mx = part[0];
#pragma unroll
    for (int e = 1; e < NEXP; e++) mx = fmaxf(mx, part[e]);
    float pe[NEXP], s = 0.f;
#pragma unroll
    for (int e = 0; e < NEXP; e++) { pe[e] = __expf(part[e] - mx); s += pe[e]; }
#pragma unroll
    for (int e = 0; e < NEXP; e++) pe[e] /= s;
    int s0 = 0;
#pragma unroll
    for (int e = 1; e < NEXP; e++) if (pe[e] > pe[s0]) s0 = e;
    int s1 = -1;
#pragma unroll
    for (int e = 0; e < NEXP; e++) if (e != s0 && (s1 < 0 || pe[e] > pe[s1])) s1 = e;
    float denom = pe[s0] + pe[s1];
    float w0 = pe[s0] / denom, w1 = pe[s1] / denom;
#pragma unroll
    for (int e = 0; e < NEXP; e++)
      cw[(size_t)row * NEXP + e] = (e == s0) ? w0 : ((e == s1) ? w1 : 0.f);
#pragma unroll
    for (int e = 0; e < NEXP; e++) atomicAdd(&accum[e], pe[e]);   // P sums
    atomicAdd(&accum[NEXP + s0], 1.0f);                           // selection counts
    atomicAdd(&accum[NEXP + s1], 1.0f);
  }
}

__global__ void init_accum_kernel(float* accum) {
  if (threadIdx.x < 16) accum[threadIdx.x] = 0.f;
}

__global__ void aux_kernel(const float* __restrict__ accum, float* __restrict__ out_aux) {
  if (threadIdx.x == 0 && blockIdx.x == 0) {
    float s = 0.f;
    for (int e = 0; e < NEXP; e++)
      s += (accum[NEXP + e] / (float)(NTOK * 2)) * (accum[e] / (float)NTOK);
    out_aux[0] = (float)NEXP * s;
  }
}

__global__ void copy4_kernel(const float4* __restrict__ src, float4* __restrict__ dst) {
  size_t i = (size_t)blockIdx.x * 256 + threadIdx.x;
  dst[i] = src[i];
}

// ---------------- launcher ----------------
extern "C" void kernel_launch(void* const* d_in, const int* in_sizes, int n_in,
                              void* d_out, int out_size, void* d_ws, size_t ws_size,
                              hipStream_t stream) {
  (void)in_sizes; (void)n_in; (void)out_size; (void)ws_size;
  const float* x   = (const float*)d_in[0];
  const float* ln1 = (const float*)d_in[1];
  const float* ln2 = (const float*)d_in[2];
  const float* Wq  = (const float*)d_in[3];
  const float* Wk  = (const float*)d_in[4];
  const float* Wv  = (const float*)d_in[5];
  const float* Wo  = (const float*)d_in[6];
  const float* Wg  = (const float*)d_in[7];
  const float* W1  = (const float*)d_in[8];
  const float* W2  = (const float*)d_in[9];
  float* outx = (float*)d_out;

  char* ws = (char*)d_ws;
  size_t off = 0;
  auto alloc = [&](size_t bytes) -> void* {
    void* p = ws + off;
    off += (bytes + 255) & ~(size_t)255;
    return p;
  };
  float*          xn1  = (float*)alloc((size_t)NTOK * CDIM * 4);
  float*          qf   = (float*)alloc((size_t)NTOK * CDIM * 4);
  float*          kf   = (float*)alloc((size_t)NTOK * 256 * 4);
  float*          vf   = (float*)alloc((size_t)NTOK * 256 * 4);
  unsigned short* qb   = (unsigned short*)alloc((size_t)NTOK * CDIM * 2);
  unsigned short* kbuf = (unsigned short*)alloc((size_t)NTOK * 256 * 2);
  unsigned short* vbuf = (unsigned short*)alloc((size_t)NTOK * 256 * 2);
  float*          yatt = (float*)alloc((size_t)NTOK * CDIM * 4);
  float*          x1   = (float*)alloc((size_t)NTOK * CDIM * 4);
  float*          xn2  = (float*)alloc((size_t)NTOK * CDIM * 4);
  unsigned short* hbf  = (unsigned short*)alloc((size_t)NTOK * IDIM * 2);
  float*          cw   = (float*)alloc((size_t)NTOK * NEXP * 4);
  float*          acc  = (float*)alloc(64);

  // 1) xn1 = rmsnorm(x)
  rmsnorm_kernel<<<NTOK, 256, 0, stream>>>(x, ln1, xn1);
  // 2) q/k/v projections
  gemm_wmma<0, float><<<dim3(CDIM / 128, NTOK / 128), 256, 0, stream>>>(
      xn1, Wq, qf, nullptr, nullptr, NTOK, CDIM, CDIM, CDIM, CDIM, CDIM, 0);
  gemm_wmma<0, float><<<dim3(256 / 128, NTOK / 128), 256, 0, stream>>>(
      xn1, Wk, kf, nullptr, nullptr, NTOK, 256, CDIM, CDIM, 256, 256, 0);
  gemm_wmma<0, float><<<dim3(256 / 128, NTOK / 128), 256, 0, stream>>>(
      xn1, Wv, vf, nullptr, nullptr, NTOK, 256, CDIM, CDIM, 256, 256, 0);
  // 3) RoPE + bf16 conversion
  rope_kernel<<<(NTOK * 16 * 32) / 256, 256, 0, stream>>>(qf, qb, 16, CDIM);
  rope_kernel<<<(NTOK * 4 * 32) / 256, 256, 0, stream>>>(kf, kbuf, 4, 256);
  cvt_bf16_kernel<<<(NTOK * 256) / 256, 256, 0, stream>>>(vf, vbuf);
  // 4) attention
  attn_kernel<<<dim3(64, 8), 256, 0, stream>>>(qb, kbuf, vbuf, yatt);
  // 5) x1 = yatt @ Wo + x
  gemm_wmma<2, float><<<dim3(CDIM / 128, NTOK / 128), 256, 0, stream>>>(
      yatt, Wo, x1, nullptr, x, NTOK, CDIM, CDIM, CDIM, CDIM, CDIM, 0);
  // 6) xn2 = rmsnorm(x1)
  rmsnorm_kernel<<<NTOK, 256, 0, stream>>>(x1, ln2, xn2);
  // 7) gating + aux stats
  init_accum_kernel<<<1, 32, 0, stream>>>(acc);
  gate_kernel<<<NTOK / 8, 256, 0, stream>>>(xn2, Wg, cw, acc);
  // 8) out = x1 (residual base for MoE accumulation)
  copy4_kernel<<<NTOK * CDIM / (256 * 4), 256, 0, stream>>>(
      (const float4*)x1, (float4*)outx);
  // 9) experts: h = gelu(xn2 @ W1[e]); out += cw[:,e] * (h @ W2[e])
  for (int e = 0; e < NEXP; e++) {
    const float* W1e = W1 + (size_t)e * CDIM * IDIM;
    const float* W2e = W2 + (size_t)e * IDIM * CDIM;
    gemm_wmma<1, float><<<dim3(IDIM / 128, NTOK / 128), 256, 0, stream>>>(
        xn2, W1e, nullptr, hbf, nullptr, NTOK, IDIM, CDIM, CDIM, IDIM, IDIM, e);
    gemm_wmma<3, unsigned short><<<dim3(CDIM / 128, NTOK / 128), 256, 0, stream>>>(
        hbf, W2e, outx, nullptr, cw, NTOK, CDIM, IDIM, IDIM, CDIM, CDIM, e);
  }
  // 10) aux loss scalar
  aux_kernel<<<1, 32, 0, stream>>>(acc, outx + (size_t)NTOK * CDIM);
}
